// TimeVariantRetentionLayer_37477884625132
// MI455X (gfx1250) — matrix operand order repaired and checked by reading, not compiled, not run on hardware
//
#include <hip/hip_runtime.h>
#include <hip/hip_bf16.h>

typedef __attribute__((ext_vector_type(16))) _Float16 v16h;
typedef __attribute__((ext_vector_type(8)))  float    v8f;
typedef int v4i_ __attribute__((vector_size(4 * sizeof(int))));  // matches builtin param

#define B_   2
#define L_   512
#define DIM_ 1024
#define H_   8
#define D_   16
#define M1   (B_ * L_)      // 1024 token rows
#define N1   (H_ * D_ * 8)  // 1024 qkva cols
#define K1   DIM_           // 1024
#define K2   (H_ * D_ * 2)  // 256
#define N2   DIM_           // 1024

#define WG_M 128            // 8 waves x 16 rows
#define WG_N 64             // 4 wmma subtiles of 16
#define LDS_STRIDE 40       // 32 k-floats padded to 40 (160B rows, 16B aligned)

// ---- address-space helpers (AS1 global / AS3 LDS), typed for the builtin --
__device__ __forceinline__ __attribute__((address_space(1))) v4i_*
as_global_v4(const void* p) {
  return (__attribute__((address_space(1))) v4i_*)(unsigned long long)p;
}
__device__ __forceinline__ __attribute__((address_space(3))) v4i_*
as_lds_v4(void* p) {
  return (__attribute__((address_space(3))) v4i_*)(unsigned)(unsigned long long)p;
}

#if __has_builtin(__builtin_amdgcn_global_load_async_to_lds_b128)
#define ASYNC_COPY_B128(dst, src, off) \
  __builtin_amdgcn_global_load_async_to_lds_b128(as_global_v4(src), as_lds_v4(dst), off, 0)
#else
#define ASYNC_COPY_B128(dst, src, off) \
  do { *(float4*)((char*)(dst) + (off)) = *(const float4*)((const char*)(src) + (off)); } while (0)
#endif

__device__ __forceinline__ void wait_async_done() {
#if __has_builtin(__builtin_amdgcn_s_wait_asynccnt)
  __builtin_amdgcn_s_wait_asynccnt(0);
#else
  asm volatile("s_wait_asynccnt 0x0" ::: "memory");
#endif
}

// ---------------------------------------------------------------------------
// C[M,N] = A[M,K] @ Bm[N,K]^T.  fp32 in/out, f16 WMMA math, fp32 accum.
// Workgroup: 128(M) x 64(N); 8 waves, each wave owns a 16x64 strip with 4
// accumulators so one A fragment feeds 4 v_wmma_f32_16x16x32_f16.
// B slab (64 n x 32 k) staged in LDS via GLOBAL_LOAD_ASYNC_TO_LDS_B128.
// ---------------------------------------------------------------------------
__global__ __launch_bounds__(256)
void wmma_gemm_nt(const float* __restrict__ A, const float* __restrict__ Bm,
                  float* __restrict__ C, int M, int N, int K) {
  __shared__ float Bls[WG_N * LDS_STRIDE];  // 10 KB

  const int lane = threadIdx.x & 31;
  const int wave = threadIdx.x >> 5;
  const int tilesN = N / WG_N;
  const int mt = (blockIdx.x / tilesN) * WG_M + wave * 16;
  const int nt = (blockIdx.x % tilesN) * WG_N;
  const int half = lane >> 4;   // 0: lanes 0-15, 1: lanes 16-31
  const int l16  = lane & 15;

  const float* aRow = A + (size_t)(mt + l16) * K;     // A row for this lane

  // staging assignment: thread t copies 8 contiguous k-floats of one n-row
  const int n_loc  = threadIdx.x >> 2;        // 0..63
  const int k_off  = (threadIdx.x & 3) << 3;  // 0,8,16,24
  const float* gB  = Bm + (size_t)(nt + n_loc) * K + k_off;
  float*       lB  = Bls + n_loc * LDS_STRIDE + k_off;

  v8f acc[4] = {v8f{}, v8f{}, v8f{}, v8f{}};

  for (int kt = 0; kt < K; kt += 32) {
    __syncthreads();  // previous slab fully consumed before overwrite
    ASYNC_COPY_B128(lB, gB + kt, 0);
    ASYNC_COPY_B128(lB, gB + kt, 16);
    wait_async_done();   // this wave's ASYNCcnt == 0 -> its LDS writes landed
    __syncthreads();     // all waves' writes landed

    // A fragment (ISA 16-bit A 16x32 layout), f32 -> f16 packed pairs
    v16h av;
#pragma unroll
    for (int v = 0; v < 8; ++v) {
      const int ka = ((v < 4) ? 0 : 16) + half * 8 + ((v & 3) << 1);
      const float2 a2 = *(const float2*)(aRow + kt + ka);
      av[2 * v]     = (_Float16)a2.x;
      av[2 * v + 1] = (_Float16)a2.y;
    }
    if (kt + 32 < K)
      __builtin_prefetch(aRow + kt + 32, 0, 1);  // global_prefetch_b8

    // 4 B fragments from LDS (ISA 16-bit B 32x16 layout), 4 WMMAs sharing av
#pragma unroll
    for (int s = 0; s < 4; ++s) {
      v16h bv;
#pragma unroll
      for (int v = 0; v < 8; ++v) {
        const int kb = ((v < 4) ? 0 : 16) + half * 8 + ((v & 3) << 1);
        const float2 b2 =
            *(const float2*)(Bls + (s * 16 + l16) * LDS_STRIDE + kb);
        bv[2 * v]     = (_Float16)b2.x;
        bv[2 * v + 1] = (_Float16)b2.y;
      }
      acc[s] = __builtin_amdgcn_wmma_f32_16x16x32_f16(
          false, av, false, bv, (short)0, acc[s], false, false);
    }
  }

  // C f32 16x16 layout: VGPR r -> row (r + 8*half), col = lane%16
  const int crow = mt + half * 8;
#pragma unroll
  for (int s = 0; s < 4; ++s)
#pragma unroll
    for (int r = 0; r < 8; ++r)
      C[(size_t)(crow + r) * N + (nt + s * 16 + l16)] = acc[s][r];
}

// ---------------------------------------------------------------------------
// Complex retention recurrence: h_t[o,p] = a_t[o]*h_{t-1}[o,p] + k_t[o]*v_t[p]
// hq_t[p] = sum_o q_t[o] * h_t[o,p].  One block per (b,h); thread = (p,o);
// state in registers; 16-lane shfl_xor reduction over o per step (wave32).
// ---------------------------------------------------------------------------
__global__ __launch_bounds__(256)
void retention_recurrence(const float* __restrict__ z,
                          const float* __restrict__ hid_re,
                          const float* __restrict__ hid_im,
                          float* __restrict__ hr,        // (B*L, H*D*2)
                          float* __restrict__ hid_next)  // (B,H,D,D) interleaved
{
  const int bh = blockIdx.x;          // 0..15
  const int b  = bh >> 3;
  const int h  = bh & 7;
  const int o  = threadIdx.x & 15;    // k-index (lane bits [3:0])
  const int p  = threadIdx.x >> 4;    // v-index

  const size_t hidIdx = ((size_t)(b * H_ + h) * D_ + o) * D_ + p;
  float sre = hid_re[hidIdx];
  float sim = hid_im[hidIdx];

  const size_t zBase = (size_t)b * L_ * N1 + (size_t)h * D_ * 8;
  for (int t = 0; t < L_; ++t) {
    const float* zr = z + zBase + (size_t)t * N1;
    const float qre = zr[o * 8 + 0], qim = zr[o * 8 + 1];
    const float kre = zr[o * 8 + 2], kim = zr[o * 8 + 3];
    const float vre = zr[p * 8 + 4], vim = zr[p * 8 + 5];
    float are = zr[o * 8 + 6], aim = zr[o * 8 + 7];

    // a <- a/|a| * sigmoid(log|a|^2)  ==  a * sqrt(m2)/(1+m2)
    const float m2 = are * are + aim * aim;
    const float sc = rsqrtf(m2) * m2 / (1.0f + m2);
    are *= sc; aim *= sc;

    // h = a*h + k*v   (complex)
    const float nre = are * sre - aim * sim + (kre * vre - kim * vim);
    const float nim = are * sim + aim * sre + (kre * vim + kim * vre);
    sre = nre; sim = nim;

    // contribution q*h, reduced over o within each 16-lane half-wave
    float cre = qre * sre - qim * sim;
    float cim = qre * sim + qim * sre;
#pragma unroll
    for (int mask = 1; mask < 16; mask <<= 1) {
      cre += __shfl_xor(cre, mask, 32);
      cim += __shfl_xor(cim, mask, 32);
    }
    if (o == 0) {
      float* out = hr + (size_t)(b * L_ + t) * K2 + (size_t)(h * D_ + p) * 2;
      out[0] = cre;
      out[1] = cim;
    }
  }

  // hidden_next = h_{L-1}, complex interleaved
  hid_next[hidIdx * 2 + 0] = sre;
  hid_next[hidIdx * 2 + 1] = sim;
}

// ---------------------------------------------------------------------------
extern "C" void kernel_launch(void* const* d_in, const int* in_sizes, int n_in,
                              void* d_out, int out_size, void* d_ws, size_t ws_size,
                              hipStream_t stream) {
  const float* x           = (const float*)d_in[0];  // (B,L,DIM)
  const float* hidden_real = (const float*)d_in[1];  // (B,H,D,D)
  const float* hidden_imag = (const float*)d_in[2];  // (B,H,D,D)
  const float* W_qkva      = (const float*)d_in[3];  // (H*D*8, DIM)
  const float* W_out       = (const float*)d_in[4];  // (DIM, H*D*2)

  float* y        = (float*)d_out;                    // (B,L,DIM)
  float* hid_next = y + (size_t)B_ * L_ * DIM_;       // (B,H,D,D) complex

  float* z  = (float*)d_ws;                           // (B*L, 1024)  4 MB
  float* hr = z + (size_t)M1 * N1;                    // (B*L, 256)   1 MB

  // GEMM1: z = x @ W_qkva^T   (1024 x 1024 x 1024)
  wmma_gemm_nt<<<(M1 / WG_M) * (N1 / WG_N), 256, 0, stream>>>(x, W_qkva, z,
                                                              M1, N1, K1);

  // Sequential complex retention over L, one block per (b,h)
  retention_recurrence<<<B_ * H_, 256, 0, stream>>>(z, hidden_real, hidden_imag,
                                                    hr, hid_next);

  // GEMM2: y = hr @ W_out^T   (1024 x 256 x 1024)
  wmma_gemm_nt<<<(M1 / WG_M) * (N2 / WG_N), 256, 0, stream>>>(hr, W_out, y,
                                                              M1, N2, K2);
}
